// RNNFeaturizer_31868657337093
// MI455X (gfx1250) — matrix operand order repaired and checked by reading, not compiled, not run on hardware
//
#include <hip/hip_runtime.h>
#include <hip/hip_bf16.h>
#include <stdint.h>

// Problem constants (from reference): T=256, B=128, V=256, E=128, H=1024
#define T_   256
#define B_   128
#define V_   256
#define E_   128
#define H_   1024
#define G4_  4096            // 4*H
#define NWG  64              // persistent workgroups, each owns a 16-col H slice
#define NTHR 256             // 8 waves (wave32)

typedef __attribute__((ext_vector_type(16))) __bf16       v16bf;
typedef __attribute__((ext_vector_type(8)))  float        v8f;
typedef __attribute__((ext_vector_type(4)))  unsigned int u32x4;
typedef __attribute__((ext_vector_type(8)))  unsigned int u32x8;

// fp32 -> bf16 bits (round-to-nearest-even), avoids any scalar __bf16 ops
static __device__ __forceinline__ unsigned short f2bf(float f){
  unsigned u = __builtin_bit_cast(unsigned, f);
  u = (u + 0x7FFFu + ((u >> 16) & 1u)) >> 16;
  return (unsigned short)u;
}

// Build a v16bf WMMA fragment from two 16-byte chunks (works for LDS or global)
static __device__ __forceinline__ v16bf load_frag(const unsigned short* p0,
                                                  const unsigned short* p1){
  u32x4 lo = *(const u32x4*)p0;
  u32x4 hi = *(const u32x4*)p1;
  u32x8 w;
  w[0]=lo[0]; w[1]=lo[1]; w[2]=lo[2]; w[3]=lo[3];
  w[4]=hi[0]; w[5]=hi[1]; w[6]=hi[2]; w[7]=hi[3];
  return __builtin_bit_cast(v16bf, w);
}

// Branch-free fast activations: one v_exp_f32 + one v_rcp_f32 each.
// Avoids OCML tanhf's divergent slow path and the IEEE v_div_* expansion,
static __device__ __forceinline__ float sig_fast(float x){
  return __builtin_amdgcn_rcpf(1.0f + __expf(-x));
}
static __device__ __forceinline__ float tanh_fast(float x){
  float e = __expf(-2.0f * __builtin_fabsf(x));   // in (0,1], finite for all x
  float r = (1.0f - e) * __builtin_amdgcn_rcpf(1.0f + e);
  return __builtin_copysignf(r, x);
}

// ---------------------------------------------------------------------------
// Workspace init: zero h0 ping buffer + barrier counter (ws is poisoned 0xAA)
__global__ void init_ws(unsigned int* __restrict__ h0_u32,
                        unsigned int* __restrict__ cnt){
  int i = blockIdx.x * blockDim.x + threadIdx.x;
  if (i < (B_ * H_ / 2)) h0_u32[i] = 0u;      // 128*1024 bf16 == 65536 u32
  if (i == 0) *cnt = 0u;
}

// ---------------------------------------------------------------------------
// Fold embedding + input projection + both biases:
//   G[v][g] = sum_k emb_W[v][k] * W_ih[g][k] + b_ih[g] + b_hh[g]
// V=256 makes the per-step input GEMM a pure table gather afterwards.
__global__ void build_gemb(const float* __restrict__ emb_W,
                           const float* __restrict__ W_ih,
                           const float* __restrict__ b_ih,
                           const float* __restrict__ b_hh,
                           float* __restrict__ G){
  int idx = blockIdx.x * blockDim.x + threadIdx.x;   // V_*G4_ threads
  if (idx >= V_ * G4_) return;
  int v = idx >> 12;
  int g = idx & (G4_ - 1);
  const float* e = emb_W + (size_t)v * E_;
  const float* w = W_ih  + (size_t)g * E_;
  float s = b_ih[g] + b_hh[g];
#pragma unroll 8
  for (int k = 0; k < E_; k++) s = fmaf(e[k], w[k], s);
  G[idx] = s;
}

// ---------------------------------------------------------------------------
// Persistent LSTM kernel.
//  * 64 WGs, each owns H-columns [wg*16, wg*16+16) and keeps its 64x1024 bf16
//    W_hh slice (i,f,g,o rows for those columns) in 128 KB of LDS.
//  * 8 waves per WG: wave w owns batch rows [16w, 16w+16) (M tiles).
//  * Per step: gates = h @ W_hh_slice^T via v_wmma_f32_16x16x32_bf16,
//    + G[token] gather, elementwise LSTM; c & stats live in VGPRs; h_new
//    (bf16) double-buffers through L2 with one grid barrier per step.
__global__ __launch_bounds__(NTHR)
void lstm_persist(const int*   __restrict__ tok,
                  const int*   __restrict__ seq_len,
                  const float* __restrict__ W_hh,
                  const float* __restrict__ G,
                  unsigned short* __restrict__ h0,
                  unsigned short* __restrict__ h1,
                  unsigned int* __restrict__ cnt,
                  float* __restrict__ out){
  extern __shared__ char smem[];
  unsigned short* lds_w = (unsigned short*)smem;      // [64][1024] bf16

  const int tid  = threadIdx.x;
  const int wg   = blockIdx.x;        // 0..63 : H slice
  const int hs   = wg * 16;
  const int wave = tid >> 5;          // 0..7  : batch M tile
  const int lane = tid & 31;
  const int n    = lane & 15;         // A row-in-tile / B col / D col
  const int hi   = lane >> 4;         // half-wave selector (K/M halves)

  // Stage W_hh slice into LDS as bf16. Row r = g*16 + c maps to
  // W_hh[g*H + hs + c][*].
  for (int i = tid; i < 64 * H_; i += NTHR){
    int r = i >> 10;
    int k = i & (H_ - 1);
    int g = r >> 4;
    int c = r & 15;
    lds_w[i] = f2bf(W_hh[(size_t)(g * H_ + hs + c) * H_ + k]);
  }
  __syncthreads();

  // Per-lane persistent state: 8 batch rows, M = hi*8 + r inside wave's tile
  float cs[8], cellv[8], maxcv[8], mincv[8], sumcv[8];
  int   sl[8];
#pragma unroll
  for (int r = 0; r < 8; r++){
    cs[r] = 0.f; cellv[r] = 0.f; maxcv[r] = 0.f; mincv[r] = 0.f; sumcv[r] = 0.f;
    sl[r] = seq_len[wave * 16 + hi * 8 + r];
  }

  const int arow = wave * 16 + n;     // batch row feeding this lane's A frag

  for (int t = 0; t < T_; t++){
    const unsigned short* hp = (t & 1) ? h1 : h0;   // read
    unsigned short*       hd = (t & 1) ? h0 : h1;   // write

    v8f acc[4] = {};                  // i, f, g, o accumulators (16x16 f32)

    // A-fragment ISA layout: lane holds row M=lane&15; K bytes at hi*8, hi*8+16
    const unsigned short* hrow = hp + (size_t)arow * H_ + hi * 8;
#pragma unroll 2
    for (int k0 = 0; k0 < H_; k0 += 32){
      v16bf af = load_frag(hrow + k0, hrow + k0 + 16);
#pragma unroll
      for (int g = 0; g < 4; g++){
        // B-fragment: lane holds col N=lane&15; K at hi*8, hi*8+16 (matches A)
        const unsigned short* bp = lds_w + ((g * 16 + n) * H_ + k0 + hi * 8);
        v16bf bfr = load_frag(bp, bp + 16);
        acc[g] = __builtin_amdgcn_wmma_f32_16x16x32_bf16(
            false, af, false, bfr, (short)0, acc[g], false, false);
      }
    }

    // Gather folded input projection, then elementwise LSTM + masked stats.
    // Accumulator map: VGPR r holds M = r + 8*hi, col N = lane&15.
#pragma unroll
    for (int r = 0; r < 8; r++){
      const int bb = wave * 16 + hi * 8 + r;
      const int tk = tok[t * B_ + bb];
      const float* gp = G + (size_t)tk * G4_ + hs + n;
      float gi = acc[0][r] + gp[0];
      float gf = acc[1][r] + gp[H_];
      float gg = acc[2][r] + gp[2 * H_];
      float go = acc[3][r] + gp[3 * H_];
      float cn = sig_fast(gf) * cs[r] + sig_fast(gi) * tanh_fast(gg);
      float hv = sig_fast(go) * tanh_fast(cn);
      cs[r] = cn;

      bool valid = (t < sl[r]);
      bool first = (t == 0);
      float mx = fmaxf(maxcv[r], cn);
      float mn = fminf(mincv[r], cn);
      float sm = sumcv[r] + cn;
      cellv[r] = valid ? cn : cellv[r];
      maxcv[r] = first ? cn : (valid ? mx : maxcv[r]);
      mincv[r] = first ? cn : (valid ? mn : mincv[r]);
      sumcv[r] = first ? cn : (valid ? sm : sumcv[r]);

      hd[(size_t)bb * H_ + hs + n] = f2bf(hv);  // h always advances
    }

    // ---- device-wide barrier (monotonic counter; reset by init_ws) ----
    __syncthreads();
    if (tid == 0){
      __threadfence();
      __hip_atomic_fetch_add(cnt, 1u, __ATOMIC_RELEASE, __HIP_MEMORY_SCOPE_AGENT);
      const unsigned target = (unsigned)(NWG * (t + 1));
      while (__hip_atomic_load(cnt, __ATOMIC_ACQUIRE, __HIP_MEMORY_SCOPE_AGENT) < target){
        __builtin_amdgcn_s_sleep(2);
      }
    }
    __syncthreads();
    __threadfence();   // acquire: don't read stale h from near caches
  }

  // Emit features: [cell | max | min | mean] along 4H
#pragma unroll
  for (int r = 0; r < 8; r++){
    const int bb = wave * 16 + hi * 8 + r;
    float* op = out + (size_t)bb * G4_ + hs + n;
    op[0]      = cellv[r];
    op[H_]     = maxcv[r];
    op[2 * H_] = mincv[r];
    op[3 * H_] = sumcv[r] / (float)sl[r];
  }
}

// ---------------------------------------------------------------------------
extern "C" void kernel_launch(void* const* d_in, const int* in_sizes, int n_in,
                              void* d_out, int out_size, void* d_ws, size_t ws_size,
                              hipStream_t stream){
  (void)in_sizes; (void)n_in; (void)out_size; (void)ws_size;
  const int*   input   = (const int*)  d_in[0];
  const int*   seq_len = (const int*)  d_in[1];
  const float* emb_W   = (const float*)d_in[2];
  const float* W_ih    = (const float*)d_in[3];
  const float* W_hh    = (const float*)d_in[4];
  const float* b_ih    = (const float*)d_in[5];
  const float* b_hh    = (const float*)d_in[6];
  float*       out     = (float*)d_out;

  // Workspace layout:
  //   [0, 4MB)          : G table (V x 4H f32)
  //   [4MB, 4MB+256KB)  : h ping (B x H bf16)
  //   [.., +256KB)      : h pong
  //   then              : barrier counter
  char* ws = (char*)d_ws;
  float*          Gt  = (float*)ws;
  unsigned short* h0  = (unsigned short*)(ws + (size_t)V_ * G4_ * sizeof(float));
  unsigned short* h1  = h0 + (size_t)B_ * H_;
  unsigned int*   cnt = (unsigned int*)(ws + (size_t)V_ * G4_ * sizeof(float)
                                           + 2 * (size_t)B_ * H_ * sizeof(unsigned short));

  init_ws<<<(B_ * H_ / 2 + 255) / 256, 256, 0, stream>>>((unsigned int*)h0, cnt);
  build_gemb<<<(V_ * G4_) / 256, 256, 0, stream>>>(emb_W, W_ih, b_ih, b_hh, Gt);
  lstm_persist<<<NWG, NTHR, 64 * H_ * sizeof(unsigned short), stream>>>(
      input, seq_len, W_hh, Gt, h0, h1, cnt, out);
}
